// BiMamba_19902878450389
// MI455X (gfx1250) — matrix operand order repaired and checked by reading, not compiled, not run on hardware
//
#include <hip/hip_runtime.h>
#include <math.h>
#include <stdint.h>

// ---------------- problem constants ----------------
#define D_MODEL 512
#define D_STATE 16
#define D_CONV  4
#define D_INNER 1024
#define DT_RANK 32
#define BATCH   8
#define SEQLEN  2048
#define NROWS   (BATCH * SEQLEN)        // 16384

// ---------------- WMMA types ----------------
typedef __attribute__((ext_vector_type(16))) __bf16 v16bf;
typedef __attribute__((ext_vector_type(8)))  __bf16 v8bf;
typedef __attribute__((ext_vector_type(8)))  float  v8f;

// ---------------- GEMM tiling ----------------
#define BM 128
#define BN 128
#define BK 32
#define AST 40   // padded LDS row stride (bf16 elems)
#define WST 40

// ---------------- optional gfx1250 async global->LDS copy ----------------
#if defined(__has_builtin)
#if __has_builtin(__builtin_amdgcn_global_load_async_to_lds_b128) && \
    __has_builtin(__builtin_amdgcn_s_wait_asynccnt)
#define USE_ASYNC_LDS 1
#endif
#endif

#if USE_ASYNC_LDS
typedef __attribute__((__vector_size__(16))) int v4i;
typedef __attribute__((address_space(1))) v4i* gas1_t;   // global (cuda_device) AS
typedef __attribute__((address_space(3))) v4i* las3_t;   // LDS AS
__device__ __forceinline__ void async_b128(const void* g, void* l) {
    // LDS generic addresses keep their offset in the low 32 bits (aperture in high bits)
    __builtin_amdgcn_global_load_async_to_lds_b128(
        (gas1_t)(uintptr_t)g,
        (las3_t)(uintptr_t)(unsigned)(uintptr_t)l, 0, 0);
}
#endif

// Stage one 128x32 bf16 A-tile and one 128x32 bf16 W-tile into LDS.
// 512 16B chunks per tile, 256 threads -> 2 chunks each per tile.
__device__ __forceinline__ void stage_tiles(const __bf16* __restrict__ A, long lda, long rowBase,
                                            const __bf16* __restrict__ W, long K, int colBase, int N,
                                            int k0, __bf16* Asb, __bf16* Wsb, int tid)
{
#if USE_ASYNC_LDS
    #pragma unroll
    for (int i = 0; i < 2; ++i) {
        int c   = tid + i * 256;
        int r   = c >> 2;
        int off = (c & 3) << 3;
        int n   = colBase + r; if (n >= N) n = N - 1;
        async_b128(A + (rowBase + r) * lda + k0 + off, &Asb[r * AST + off]);
        async_b128(W + (long)n * K + k0 + off,         &Wsb[r * WST + off]);
    }
#else
    int  r[2], off[2], n[2];
    uint4 va[2], vw[2];
    #pragma unroll
    for (int i = 0; i < 2; ++i) {          // issue all 4 global loads first
        int c  = tid + i * 256;
        r[i]   = c >> 2;
        off[i] = (c & 3) << 3;
        n[i]   = colBase + r[i]; if (n[i] >= N) n[i] = N - 1;
        va[i] = *(const uint4*)(A + (rowBase + r[i]) * lda + k0 + off[i]);
        vw[i] = *(const uint4*)(W + (long)n[i] * K + k0 + off[i]);
    }
    #pragma unroll
    for (int i = 0; i < 2; ++i) {          // then the LDS stores
        *(uint4*)&Asb[r[i] * AST + off[i]] = va[i];
        *(uint4*)&Wsb[r[i] * WST + off[i]] = vw[i];
    }
#endif
}

// C[M,N] = A[M,K] @ W[N,K]^T (+bias, +softplus). bf16 operands, f32 accum, TC output.
template <typename TC>
__global__ __launch_bounds__(256)
void wmma_gemm(const __bf16* __restrict__ A, int lda,
               const __bf16* __restrict__ W,        // (N,K) row-major
               TC* __restrict__ C, int ldc,
               int M, int N, int K,
               const float* __restrict__ bias, int act)
{
    __shared__ __bf16 As[2][BM * AST];
    __shared__ __bf16 Ws[2][BN * WST];

    const int tid  = threadIdx.x;
    const int lane = tid & 31;
    const int wave = tid >> 5;
    const int wm   = (wave & 1) * 64;   // 2 waves along M
    const int wn   = (wave >> 1) * 32;  // 4 waves along N

    const long rowBase = (long)blockIdx.y * BM;
    const int  colBase = blockIdx.x * BN;

    v8f acc[4][2];
    for (int tm = 0; tm < 4; ++tm)
        for (int tn = 0; tn < 2; ++tn)
            for (int j = 0; j < 8; ++j)
                acc[tm][tn][j] = 0.0f;

    int cur = 0;
    stage_tiles(A, lda, rowBase, W, K, colBase, N, 0, As[0], Ws[0], tid);

    for (int k0 = 0; k0 < K; k0 += BK) {
        const bool hasNext = (k0 + BK) < K;
        if (hasNext)
            stage_tiles(A, lda, rowBase, W, K, colBase, N, k0 + BK,
                        As[cur ^ 1], Ws[cur ^ 1], tid);
#if USE_ASYNC_LDS
        if (hasNext) __builtin_amdgcn_s_wait_asynccnt(4);  // only the 4 newest outstanding
        else         __builtin_amdgcn_s_wait_asynccnt(0);
#endif
        __syncthreads();

        // ---- fragments per ISA 7.12.2 ----
        const __bf16* Asb = As[cur];
        const __bf16* Wsb = Ws[cur];
        v16bf afrag[4];
        #pragma unroll
        for (int tm = 0; tm < 4; ++tm) {
            int r  = wm + tm * 16 + (lane & 15);
            int kh = (lane >> 4) * 8;     // lanes<16: K{0..7,16..23}; >=16: {8..15,24..31}
            union { v16bf v; v8bf h[2]; } u;
            u.h[0] = *(const v8bf*)&Asb[r * AST + kh];
            u.h[1] = *(const v8bf*)&Asb[r * AST + 16 + kh];
            afrag[tm] = u.v;
        }
        v16bf bfrag[2];
        #pragma unroll
        for (int tn = 0; tn < 2; ++tn) {
            int r  = wn + tn * 16 + (lane & 15);  // output column = W row
            int kh = (lane >> 4) * 8;             // lanes<16: K{0..7,16..23}; >=16: {8..15,24..31}
            union { v16bf v; v8bf h[2]; } u;
            u.h[0] = *(const v8bf*)&Wsb[r * WST + kh];
            u.h[1] = *(const v8bf*)&Wsb[r * WST + 16 + kh];
            bfrag[tn] = u.v;
        }
        #pragma unroll
        for (int tm = 0; tm < 4; ++tm)
            #pragma unroll
            for (int tn = 0; tn < 2; ++tn)
                acc[tm][tn] = __builtin_amdgcn_wmma_f32_16x16x32_bf16(
                    false, afrag[tm], false, bfrag[tn],
                    (short)0, acc[tm][tn], false, false);
        __syncthreads();   // protect buffer about to be overwritten
        cur ^= 1;
    }

    // ---- epilogue: VGPR j -> M=j (lanes<16) / M=j+8 (lanes>=16) ----
    for (int tm = 0; tm < 4; ++tm)
        for (int tn = 0; tn < 2; ++tn) {
            int n = colBase + wn + tn * 16 + (lane & 15);
            if (n < N) {
                long m0 = rowBase + wm + tm * 16 + (lane >> 4) * 8;
                float bv = bias ? bias[n] : 0.0f;
                #pragma unroll
                for (int j = 0; j < 8; ++j) {
                    float v = acc[tm][tn][j] + bv;
                    if (act == 1) v = (v > 20.0f) ? v : log1pf(__expf(v));
                    C[(m0 + j) * (long)ldc + n] = (TC)v;
                }
            }
        }
}

// ---------------- fp32 -> bf16 conversion ----------------
__global__ __launch_bounds__(256)
void cvt_bf16_kernel(const float* __restrict__ in, __bf16* __restrict__ outp, long n)
{
    long i = (long)blockIdx.x * blockDim.x + threadIdx.x;
    if (i < n) outp[i] = (__bf16)in[i];
}

// ---------------- causal depthwise conv (D_CONV=4) + bias + SiLU ----------------
__global__ __launch_bounds__(256)
void conv_silu_kernel(const __bf16* __restrict__ xz,   // (NROWS, 2048), xi = cols 0..1023
                      const float* __restrict__ cw,    // (D_INNER, 4)
                      const float* __restrict__ cb,    // (D_INNER)
                      __bf16* __restrict__ xi,         // (NROWS, D_INNER)
                      int dir)
{
    long idx = (long)blockIdx.x * blockDim.x + threadIdx.x;
    if (idx >= (long)NROWS * D_INNER) return;
    int  d     = (int)(idx & (D_INNER - 1));
    long row   = idx >> 10;
    int  t     = (int)(row & (SEQLEN - 1));
    long bbase = row - t;                  // b * SEQLEN
    float acc = cb[d];
    #pragma unroll
    for (int j = 0; j < D_CONV; ++j) {
        int tt = (dir == 0) ? (t - 3 + j) : (t + 3 - j);
        if (tt >= 0 && tt < SEQLEN)
            acc += cw[d * D_CONV + j] * (float)xz[(bbase + tt) * (2 * D_INNER) + d];
    }
    xi[idx] = (__bf16)(acc / (1.0f + __expf(-acc))); // SiLU
}

// ---------------- selective scan: serial in t, parallel over (batch, d) ----------------
__global__ __launch_bounds__(256)
void scan_kernel(const __bf16* __restrict__ dblb,  // (NROWS, 64): [dt_rank | B(16) | C(16)]
                 const __bf16* __restrict__ dt,    // (NROWS, D_INNER) post-softplus
                 const __bf16* __restrict__ xi,    // (NROWS, D_INNER)
                 const float*  __restrict__ A_log, // (D_INNER, 16)
                 __bf16* __restrict__ ys,          // (NROWS, D_INNER)
                 int dir)
{
    int d = blockIdx.x * 256 + threadIdx.x;   // gridDim.x = 4
    int b = blockIdx.y;
    float Ac[D_STATE];
    #pragma unroll
    for (int n = 0; n < D_STATE; ++n) Ac[n] = -__expf(A_log[d * D_STATE + n]);
    float h[D_STATE];
    #pragma unroll
    for (int n = 0; n < D_STATE; ++n) h[n] = 0.0f;

    long base = (long)b * SEQLEN;
    for (int s = 0; s < SEQLEN; ++s) {
        int  t = (dir == 0) ? s : (SEQLEN - 1 - s);
        long r = base + t;
        // B/C are uniform per block per step: load as packed dwords -> scalar loads
        const unsigned* bc = (const unsigned*)(dblb + r * 64 + DT_RANK);
        float Bv[D_STATE], Cv[D_STATE];
        #pragma unroll
        for (int q = 0; q < 8; ++q) {
            union { unsigned u; __bf16 h2[2]; } ub, uc;
            ub.u = bc[q]; uc.u = bc[8 + q];
            Bv[2*q]   = (float)ub.h2[0]; Bv[2*q+1] = (float)ub.h2[1];
            Cv[2*q]   = (float)uc.h2[0]; Cv[2*q+1] = (float)uc.h2[1];
        }
        float dtv = (float)dt[r * D_INNER + d];
        float xv  = (float)xi[r * D_INNER + d];
        float dx  = dtv * xv;
        float y   = 0.0f;
        #pragma unroll
        for (int n = 0; n < D_STATE; ++n) {
            h[n] = __expf(dtv * Ac[n]) * h[n] + dx * Bv[n];
            y   += h[n] * Cv[n];
        }
        ys[r * D_INNER + d] = (__bf16)y;
    }
}

// ---------------- gating: ys = (ys + xi*Dp) * silu(z) ----------------
__global__ __launch_bounds__(256)
void gate_kernel(__bf16* __restrict__ ys, const __bf16* __restrict__ xi,
                 const __bf16* __restrict__ xz, const float* __restrict__ Dp)
{
    long idx = (long)blockIdx.x * blockDim.x + threadIdx.x;
    if (idx >= (long)NROWS * D_INNER) return;
    int  d   = (int)(idx & (D_INNER - 1));
    long row = idx >> 10;
    float z  = (float)xz[row * (2 * D_INNER) + D_INNER + d];
    float sz = z / (1.0f + __expf(-z));
    ys[idx]  = (__bf16)(((float)ys[idx] + (float)xi[idx] * Dp[d]) * sz);
}

// ---------------- launchers ----------------
template <typename TC>
static inline void launch_gemm(const __bf16* A, int lda, const __bf16* W,
                               TC* C, int ldc, int M, int N, int K,
                               const float* bias, int act, hipStream_t s)
{
    dim3 grid((N + BN - 1) / BN, M / BM);
    wmma_gemm<TC><<<grid, 256, 0, s>>>(A, lda, W, C, ldc, M, N, K, bias, act);
}

static inline void launch_cvt(const float* in, __bf16* outp, long n, hipStream_t s)
{
    cvt_bf16_kernel<<<(int)((n + 255) / 256), 256, 0, s>>>(in, outp, n);
}

extern "C" void kernel_launch(void* const* d_in, const int* in_sizes, int n_in,
                              void* d_out, int out_size, void* d_ws, size_t ws_size,
                              hipStream_t stream)
{
    const float* xf    = (const float*)d_in[0];
    const float* lin_w = (const float*)d_in[19];
    const float* lin_b = (const float*)d_in[20];
    float* out = (float*)d_out;

    // ---- workspace layout (bf16 elements; all region sizes multiples of 8 -> 16B aligned)
    __bf16* p = (__bf16*)d_ws;
    __bf16* xbf   = p; p += (size_t)NROWS * D_MODEL;        // x in bf16
    __bf16* w_in [2]; __bf16* w_xp[2]; __bf16* w_dt[2]; __bf16* w_out[2];
    for (int d = 0; d < 2; ++d) {
        w_in [d] = p; p += (size_t)2 * D_INNER * D_MODEL;   // 2048*512
        w_xp [d] = p; p += (size_t)(DT_RANK + 2 * D_STATE) * D_INNER;
        w_dt [d] = p; p += (size_t)D_INNER * DT_RANK;
        w_out[d] = p; p += (size_t)D_MODEL * D_INNER;
    }
    __bf16* w_lin = p; p += (size_t)D_MODEL * 2 * D_MODEL;
    __bf16* xz   = p; p += (size_t)NROWS * 2 * D_INNER;
    __bf16* xi   = p; p += (size_t)NROWS * D_INNER;
    __bf16* dbl  = p; p += (size_t)NROWS * 64;
    __bf16* dtb  = p; p += (size_t)NROWS * D_INNER;
    __bf16* ys   = p; p += (size_t)NROWS * D_INNER;
    __bf16* ycat = p; p += (size_t)NROWS * D_INNER;

    // ---- one-time fp32 -> bf16 conversions (x + all GEMM weights)
    launch_cvt(xf, xbf, (long)NROWS * D_MODEL, stream);
    for (int d = 0; d < 2; ++d) {
        int base = 1 + d * 9;
        launch_cvt((const float*)d_in[base + 0], w_in [d], (long)2 * D_INNER * D_MODEL, stream);
        launch_cvt((const float*)d_in[base + 3], w_xp [d], (long)(DT_RANK + 2 * D_STATE) * D_INNER, stream);
        launch_cvt((const float*)d_in[base + 4], w_dt [d], (long)D_INNER * DT_RANK, stream);
        launch_cvt((const float*)d_in[base + 8], w_out[d], (long)D_MODEL * D_INNER, stream);
    }
    launch_cvt(lin_w, w_lin, (long)D_MODEL * 2 * D_MODEL, stream);

    const long elemN  = (long)NROWS * D_INNER;
    const int  blocksE = (int)((elemN + 255) / 256);

    for (int dir = 0; dir < 2; ++dir) {
        int base = 1 + dir * 9;
        const float* conv_w = (const float*)d_in[base + 1];
        const float* conv_b = (const float*)d_in[base + 2];
        const float* dt_b   = (const float*)d_in[base + 5];
        const float* A_log  = (const float*)d_in[base + 6];
        const float* Dp     = (const float*)d_in[base + 7];

        // 1) xz = x @ in_w^T              (16384 x 2048 x 512)
        launch_gemm<__bf16>(xbf, D_MODEL, w_in[dir], xz, 2 * D_INNER,
                            NROWS, 2 * D_INNER, D_MODEL, nullptr, 0, stream);
        // 2) depthwise causal conv + SiLU (direction-dependent window)
        conv_silu_kernel<<<blocksE, 256, 0, stream>>>(xz, conv_w, conv_b, xi, dir);
        // 3) dbl = xi @ xp_w^T            (16384 x 64 x 1024)
        launch_gemm<__bf16>(xi, D_INNER, w_xp[dir], dbl, 64,
                            NROWS, DT_RANK + 2 * D_STATE, D_INNER, nullptr, 0, stream);
        // 4) dt = softplus(dbl[:, :32] @ dt_w^T + dt_b)
        launch_gemm<__bf16>(dbl, 64, w_dt[dir], dtb, D_INNER,
                            NROWS, D_INNER, DT_RANK, dt_b, 1, stream);
        // 5) selective scan (serial over time, direction-dependent traversal)
        dim3 sg(D_INNER / 256, BATCH);
        scan_kernel<<<sg, 256, 0, stream>>>(dbl, dtb, xi, A_log, ys, dir);
        // 6) gate: ys = (ys + xi*Dp) * silu(z)
        gate_kernel<<<blocksE, 256, 0, stream>>>(ys, xi, xz, Dp);
        // 7) ycat[:, dir*512 : (dir+1)*512] = ys @ out_w^T
        launch_gemm<__bf16>(ys, D_INNER, w_out[dir], ycat + dir * D_MODEL, 2 * D_MODEL,
                            NROWS, D_MODEL, D_INNER, nullptr, 0, stream);
    }

    // 8) out = ycat @ lin_w^T + lin_b     (16384 x 512 x 1024), fp32 output
    launch_gemm<float>(ycat, 2 * D_MODEL, w_lin, out, D_MODEL,
                       NROWS, D_MODEL, 2 * D_MODEL, lin_b, 0, stream);
}
